// network_fusion_74938589380968
// MI455X (gfx1250) — matrix operand order repaired and checked by reading, not compiled, not run on hardware
//
#include <hip/hip_runtime.h>
#include <math.h>

// ---------------------------------------------------------------------------
// VMamba-style dual-branch 2D selective scan block for gfx1250 (MI455X).
// GEMMs use v_wmma_f32_16x16x32_f16 (wave32 WMMA). The sequential selective
// scan is parallelized over 1024 independent recurrences (2 per wave32,
// 16 state-lanes each) with shfl_xor reductions per timestep.
// ---------------------------------------------------------------------------

typedef _Float16 half_t;
typedef __attribute__((ext_vector_type(16))) _Float16 v16h;
typedef __attribute__((ext_vector_type(8)))  _Float16 v8h;
typedef __attribute__((ext_vector_type(8)))  float    v8f;

#define D_MODEL 64
#define D_STATE 16
#define D_INNER 128
#define DT_RANK 4
#define KDIRS   4
#define HDIM    64
#define WDIM    64
#define LSEQ    4096   // 64*64

// ---------------------------------------------------------------------------
// Direction index mapping (cross-scan):
//  k=0 : row-major flatten           k=1 : transpose(H,W) flatten
//  k=2 : flip of k0                  k=3 : flip of k1
// Returns pixel p = h*64+w in the canonical (H,W) layout for scan index l.
// ---------------------------------------------------------------------------
__device__ __forceinline__ int pix_map(int k, int l) {
  int l2 = (k & 2) ? (LSEQ - 1 - l) : l;
  int h, w;
  if (k & 1) { w = l2 >> 6; h = l2 & 63; }
  else       { h = l2 >> 6; w = l2 & 63; }
  return (h << 6) + w;
}

// ---------------------------------------------------------------------------
// WMMA fragment helpers (layouts per CDNA5 ISA 7.12.2, f16 16x16x32)
// A (16xK=32): lane m = lane&15; elem j -> K = (j<8 ? j : j+8) + 8*(lane>>4)
//   => two contiguous 8-half (16B) chunks per lane.
// B (K=32 x 16): lane n = lane&15; elem j -> K = (j<8 ? j : j+8) + 8*(lane>>4)
//   => pre-packed so each fragment is one contiguous 32B load.
// C/D f32: lane n = lane&15; elem i -> M = i + 8*(lane>>4)
// ---------------------------------------------------------------------------
__device__ __forceinline__ v16h load_a_frag(const half_t* A, int lda,
                                            int m0, int k0, int lane) {
  int m  = m0 + (lane & 15);
  int hi = lane >> 4;
  const half_t* row = A + (size_t)m * lda + k0 + 8 * hi;
  v8h lo = *(const v8h*)(row);
  v8h up = *(const v8h*)(row + 16);
  v16h a;
#pragma unroll
  for (int j = 0; j < 8; ++j) { a[j] = lo[j]; a[j + 8] = up[j]; }
  return a;
}

__device__ __forceinline__ v8f wmma_acc(v16h a, v16h b, v8f c) {
  return __builtin_amdgcn_wmma_f32_16x16x32_f16(
      /*neg_a=*/false, a, /*neg_b=*/false, b,
      /*c_mod=*/(short)0, c, /*reuse_a=*/false, /*reuse_b=*/false);
}

// ---------------------------------------------------------------------------
// Pack an (N x Kdim) row-major f32 weight into WMMA B fragments:
// packed[(((ks*nTiles + nt)*32 + lane)*16 + j)] = W[n][k],
//   n = nt*16 + (lane&15),  k = ks*32 + 8*(lane>>4) + (j<8 ? j : j+8); zero-pad OOB.
// ---------------------------------------------------------------------------
__global__ void pack_b_kernel(const float* __restrict__ W, half_t* __restrict__ out,
                              int Kdim, int Nvalid, int kSteps, int nTiles) {
  int idx = blockIdx.x * blockDim.x + threadIdx.x;
  int total = kSteps * nTiles * 512;
  if (idx >= total) return;
  int j    = idx & 15;
  int lane = (idx >> 4) & 31;
  int tile = idx >> 9;
  int nt   = tile % nTiles;
  int ks   = tile / nTiles;
  int n    = nt * 16 + (lane & 15);
  int kk   = ks * 32 + 8 * (lane >> 4) + (j < 8 ? j : j + 8);
  float v = (n < Nvalid && kk < Kdim) ? W[(size_t)n * Kdim + kk] : 0.f;
  out[idx] = (half_t)v;
}

// Convert x0/x1 (L x 64 f32) -> f16 row-major
__global__ void cvt_x_kernel(const float* __restrict__ x0, const float* __restrict__ x1,
                             half_t* __restrict__ xh) {
  int idx = blockIdx.x * blockDim.x + threadIdx.x;
  if (idx >= 2 * LSEQ * D_MODEL) return;
  int s = idx >> 18;  // 2^18 = 4096*64
  int r = idx & (LSEQ * D_MODEL - 1);
  const float* src = s ? x1 : x0;
  xh[idx] = (half_t)src[r];
}

// ---------------------------------------------------------------------------
// in_proj GEMM: out(4096 x 256) = Xh(4096 x 64) @ W^T(64 x 256)
// n<128 -> xi in NCHW ([d][l]) for the conv; n>=128 -> z in [l][128].
// ---------------------------------------------------------------------------
__global__ void inproj_gemm_kernel(const half_t* __restrict__ Xh,
                                   const half_t* __restrict__ Bp,
                                   float* __restrict__ xin, float* __restrict__ z) {
  int wave = (blockIdx.x * blockDim.x + threadIdx.x) >> 5;
  int lane = threadIdx.x & 31;
  if (wave >= 256 * 16) return;
  int mt = wave >> 4, nt = wave & 15;
  v8f c = {};
#pragma unroll
  for (int ks = 0; ks < 2; ++ks) {
    v16h a = load_a_frag(Xh, D_MODEL, mt * 16, ks * 32, lane);
    v16h b = *(const v16h*)(Bp + ((size_t)(ks * 16 + nt) * 32 + lane) * 16);
    c = wmma_acc(a, b, c);
  }
  int n = nt * 16 + (lane & 15);
  int mbase = mt * 16 + ((lane >> 4) << 3);
#pragma unroll
  for (int i = 0; i < 8; ++i) {
    int m = mbase + i;
    float v = c[i];
    if (n < D_INNER) xin[(size_t)n * LSEQ + m] = v;
    else             z[(size_t)m * D_INNER + (n - D_INNER)] = v;
  }
}

// Depthwise 3x3 conv (SAME, groups=128) + bias + SiLU.  xin/xc: [128][4096]
__global__ void dwconv_silu_kernel(const float* __restrict__ xin,
                                   const float* __restrict__ cw,
                                   const float* __restrict__ cb,
                                   float* __restrict__ xc) {
  int idx = blockIdx.x * blockDim.x + threadIdx.x;
  if (idx >= D_INNER * LSEQ) return;
  int d = idx >> 12, l = idx & (LSEQ - 1);
  int h = l >> 6, w = l & 63;
  float acc = cb[d];
#pragma unroll
  for (int dy = -1; dy <= 1; ++dy) {
#pragma unroll
    for (int dx = -1; dx <= 1; ++dx) {
      int hh = h + dy, ww = w + dx;
      if (hh >= 0 && hh < HDIM && ww >= 0 && ww < WDIM)
        acc += cw[d * 9 + (dy + 1) * 3 + (dx + 1)] *
               xin[(size_t)d * LSEQ + (hh << 6) + ww];
    }
  }
  xc[idx] = acc / (1.f + __expf(-acc));  // SiLU
}

// Build xs0^T f16 [k][l][128] (A operand for the x_proj GEMM, branch 0 only)
__global__ void pack_xsT_kernel(const float* __restrict__ xc0, half_t* __restrict__ xsT) {
  int idx = blockIdx.x * blockDim.x + threadIdx.x;
  if (idx >= KDIRS * LSEQ * D_INNER) return;
  int d = idx & 127;
  int l = (idx >> 7) & (LSEQ - 1);
  int k = idx >> 19;
  xsT[idx] = (half_t)xc0[(size_t)d * LSEQ + pix_map(k, l)];
}

// ---------------------------------------------------------------------------
// x_proj GEMM per direction: xdbl[k](4096 x 48) = xs0T[k](4096x128) @ Wx^T(128x48)
// cols 0..3 = dt_rank, 4..19 = B, 20..35 = C, 36..47 zero pad.
// ---------------------------------------------------------------------------
__global__ void xdbl_gemm_kernel(const half_t* __restrict__ xsT,
                                 const half_t* __restrict__ BpX,
                                 float* __restrict__ xdbl) {
  int wave = (blockIdx.x * blockDim.x + threadIdx.x) >> 5;
  int lane = threadIdx.x & 31;
  if (wave >= KDIRS * 256 * 3) return;
  int k  = wave / (256 * 3);
  int r  = wave % (256 * 3);
  int mt = r / 3, nt = r % 3;
  const half_t* A  = xsT + (size_t)k * LSEQ * D_INNER;
  const half_t* Bp = BpX + (size_t)k * 4 * 3 * 512;
  v8f c = {};
#pragma unroll
  for (int ks = 0; ks < 4; ++ks) {
    v16h a = load_a_frag(A, D_INNER, mt * 16, ks * 32, lane);
    v16h b = *(const v16h*)(Bp + ((size_t)(ks * 3 + nt) * 32 + lane) * 16);
    c = wmma_acc(a, b, c);
  }
  int n = nt * 16 + (lane & 15);
  int mbase = mt * 16 + ((lane >> 4) << 3);
#pragma unroll
  for (int i = 0; i < 8; ++i)
    xdbl[((size_t)k * LSEQ + mbase + i) * 48 + n] = c[i];
}

// delta[s][k][d][l] = softplus( dt_w_s[k,d,:] . xdbl[k,l,0:4] + dt_b_s[k,d] )
__global__ void delta_kernel(const float* __restrict__ xdbl,
                             const float* __restrict__ dt_w0, const float* __restrict__ dt_w1,
                             const float* __restrict__ dt_b0, const float* __restrict__ dt_b1,
                             float* __restrict__ delta) {
  int idx = blockIdx.x * blockDim.x + threadIdx.x;
  if (idx >= 2 * KDIRS * D_INNER * LSEQ) return;
  int l = idx & (LSEQ - 1);
  int d = (idx >> 12) & 127;
  int k = (idx >> 19) & 3;
  int s = idx >> 21;
  const float* dtw = (s ? dt_w1 : dt_w0) + (size_t)(k * D_INNER + d) * DT_RANK;
  float bias       = (s ? dt_b1 : dt_b0)[k * D_INNER + d];
  const float* xr  = xdbl + ((size_t)k * LSEQ + l) * 48;
  float x = dtw[0]*xr[0] + dtw[1]*xr[1] + dtw[2]*xr[2] + dtw[3]*xr[3] + bias;
  delta[idx] = (x > 20.f) ? x : log1pf(__expf(x));
}

// ---------------------------------------------------------------------------
// Selective scan: 1024 independent recurrences (s in {0,1}, k in 0..3, d in
// 0..127), 2 per wave32, 16 state-lanes each.  Cross-branch parameters:
//   s=0: u = xs0 (from xc0), delta/A/D from set 1;  s=1: u = xs1, set 0.
// B and C always come from branch-0's xdbl (faithful to reference).
// ---------------------------------------------------------------------------
__global__ void scan_kernel(const float* __restrict__ delta,
                            const float* __restrict__ xdbl,
                            const float* __restrict__ xc0, const float* __restrict__ xc1,
                            const float* __restrict__ Alog0, const float* __restrict__ Alog1,
                            const float* __restrict__ Ds0, const float* __restrict__ Ds1,
                            float* __restrict__ outy) {
  int wave = (blockIdx.x * blockDim.x + threadIdx.x) >> 5;
  int lane = threadIdx.x & 31;
  int sc = wave * 2 + (lane >> 4);
  if (sc >= 1024) return;
  int n = lane & 15;
  int s = sc >> 9;
  int rem = sc & 511;
  int k = rem >> 7, d = rem & 127;

  const float* Alog = (s == 0) ? Alog1 : Alog0;
  const float* Dp   = (s == 0) ? Ds1   : Ds0;
  float A  = -__expf(Alog[((k * D_INNER + d) << 4) + n]);
  float Dv = Dp[k * D_INNER + d];

  const float* dptr = delta + ((size_t)(((1 - s) * KDIRS + k) * D_INNER + d) << 12);
  const float* u_ch = ((s == 0) ? xc0 : xc1) + ((size_t)d << 12);
  const float* bc   = xdbl + (size_t)k * LSEQ * 48;
  float* yo         = outy + ((size_t)((s * KDIRS + k) * D_INNER + d) << 12);

  float h = 0.f;
  for (int l = 0; l < LSEQ; ++l) {
    __builtin_prefetch(dptr + l + 64, 0, 1);   // global_prefetch_b8
    float dt = dptr[l];
    float u  = u_ch[pix_map(k, l)];
    const float* row = bc + (size_t)l * 48;
    float Bv = row[4 + n];
    float Cv = row[20 + n];
    float da = __expf(dt * A);
    h = da * h + dt * Bv * u;
    float py = h * Cv;
    // reduce the 16 state lanes (stays within each 16-lane group)
    py += __shfl_xor(py, 1, 32);
    py += __shfl_xor(py, 2, 32);
    py += __shfl_xor(py, 4, 32);
    py += __shfl_xor(py, 8, 32);
    if (n == 0) yo[l] = py + u * Dv;
  }
}

// Merge 4 directions -> ym[s][l][128] (pixel-major for layernorm)
__global__ void merge_kernel(const float* __restrict__ outy, float* __restrict__ ym) {
  int idx = blockIdx.x * blockDim.x + threadIdx.x;
  if (idx >= 2 * LSEQ * D_INNER) return;
  int d = idx & 127;
  int l = (idx >> 7) & (LSEQ - 1);
  int s = idx >> 19;
  int h = l >> 6, w = l & 63;
  int lt = (w << 6) + h;
  const float* base = outy + (size_t)s * KDIRS * D_INNER * LSEQ + (size_t)d * LSEQ;
  const size_t kst = (size_t)D_INNER * LSEQ;
  float v = base[0 * kst + l]
          + base[2 * kst + (LSEQ - 1 - l)]
          + base[1 * kst + lt]
          + base[3 * kst + (LSEQ - 1 - lt)];
  ym[((size_t)s * LSEQ + l) * D_INNER + d] = v;
}

// LayerNorm(128) + SiLU(z) gate, f16 output for the out_proj GEMM.
// One wave per (s, pixel); each lane handles 4 channels.
__global__ void ln_gate_kernel(const float* __restrict__ ym,
                               const float* __restrict__ z0, const float* __restrict__ z1,
                               const float* __restrict__ g0, const float* __restrict__ b0,
                               const float* __restrict__ g1, const float* __restrict__ b1,
                               half_t* __restrict__ gy) {
  int wave = (blockIdx.x * blockDim.x + threadIdx.x) >> 5;
  int lane = threadIdx.x & 31;
  if (wave >= 2 * LSEQ) return;
  int s = wave >> 12;
  int l = wave & (LSEQ - 1);
  const float* yrow = ym + ((size_t)s * LSEQ + l) * D_INNER;
  float v[4], sum = 0.f;
#pragma unroll
  for (int j = 0; j < 4; ++j) { v[j] = yrow[lane * 4 + j]; sum += v[j]; }
#pragma unroll
  for (int off = 1; off < 32; off <<= 1) sum += __shfl_xor(sum, off, 32);
  float mean = sum * (1.f / 128.f);
  float sq = 0.f;
#pragma unroll
  for (int j = 0; j < 4; ++j) { float t = v[j] - mean; sq += t * t; }
#pragma unroll
  for (int off = 1; off < 32; off <<= 1) sq += __shfl_xor(sq, off, 32);
  float rstd = rsqrtf(sq * (1.f / 128.f) + 1e-5f);
  const float* g = s ? g1 : g0;
  const float* b = s ? b1 : b0;
  const float* zrow = (s ? z1 : z0) + (size_t)l * D_INNER;
#pragma unroll
  for (int j = 0; j < 4; ++j) {
    int c = lane * 4 + j;
    float zn = zrow[c];
    float gate = zn / (1.f + __expf(-zn));
    float out = ((v[j] - mean) * rstd * g[c] + b[c]) * gate;
    gy[((size_t)s * LSEQ + l) * D_INNER + c] = (half_t)out;
  }
}

// out_proj GEMM: out[s](4096 x 64) = gy[s](4096x128) @ Wout^T(128x64), f32 out
__global__ void outproj_gemm_kernel(const half_t* __restrict__ gy,
                                    const half_t* __restrict__ BpO,
                                    float* __restrict__ out) {
  int wave = (blockIdx.x * blockDim.x + threadIdx.x) >> 5;
  int lane = threadIdx.x & 31;
  if (wave >= 2 * 256 * 4) return;
  int s  = wave / (256 * 4);
  int r  = wave % (256 * 4);
  int mt = r >> 2, nt = r & 3;
  const half_t* A  = gy  + (size_t)s * LSEQ * D_INNER;
  const half_t* Bp = BpO + (size_t)s * 4 * 4 * 512;
  v8f c = {};
#pragma unroll
  for (int ks = 0; ks < 4; ++ks) {
    v16h a = load_a_frag(A, D_INNER, mt * 16, ks * 32, lane);
    v16h b = *(const v16h*)(Bp + ((size_t)(ks * 4 + nt) * 32 + lane) * 16);
    c = wmma_acc(a, b, c);
  }
  int n = nt * 16 + (lane & 15);
  int mbase = mt * 16 + ((lane >> 4) << 3);
#pragma unroll
  for (int i = 0; i < 8; ++i)
    out[(size_t)s * LSEQ * D_MODEL + (size_t)(mbase + i) * D_MODEL + n] = c[i];
}

// ---------------------------------------------------------------------------
// Host launch
// ---------------------------------------------------------------------------
static inline int cdiv(int a, int b) { return (a + b - 1) / b; }

extern "C" void kernel_launch(void* const* d_in, const int* in_sizes, int n_in,
                              void* d_out, int out_size, void* d_ws, size_t ws_size,
                              hipStream_t stream) {
  (void)in_sizes; (void)n_in; (void)out_size; (void)ws_size;
  const float* x0        = (const float*)d_in[0];
  const float* x1        = (const float*)d_in[1];
  const float* in_w0     = (const float*)d_in[2];
  const float* in_w1     = (const float*)d_in[3];
  const float* conv0_w   = (const float*)d_in[4];
  const float* conv0_b   = (const float*)d_in[5];
  const float* conv1_w   = (const float*)d_in[6];
  const float* conv1_b   = (const float*)d_in[7];
  const float* xproj_w0  = (const float*)d_in[8];
  // d_in[9] (x_proj_w1) intentionally unused (faithful to reference)
  const float* dt_w0     = (const float*)d_in[10];
  const float* dt_w1     = (const float*)d_in[11];
  const float* dt_b0     = (const float*)d_in[12];
  const float* dt_b1     = (const float*)d_in[13];
  const float* A_logs0   = (const float*)d_in[14];
  const float* A_logs1   = (const float*)d_in[15];
  const float* Ds0       = (const float*)d_in[16];
  const float* Ds1       = (const float*)d_in[17];
  const float* ln0_g     = (const float*)d_in[18];
  const float* ln0_b     = (const float*)d_in[19];
  const float* ln1_g     = (const float*)d_in[20];
  const float* ln1_b     = (const float*)d_in[21];
  const float* out_w0    = (const float*)d_in[22];
  const float* out_w1    = (const float*)d_in[23];
  float* out = (float*)d_out;

  // ---- carve workspace (total ~58.1 MB, 256B aligned chunks) ----
  char* p = (char*)d_ws;
  auto carve = [&](size_t bytes) -> char* {
    char* r = p;
    p += (bytes + 255) & ~(size_t)255;
    return r;
  };
  half_t* xh      = (half_t*)carve(2 * LSEQ * D_MODEL * sizeof(half_t));       // 1 MB
  half_t* packIn0 = (half_t*)carve(2 * 16 * 512 * sizeof(half_t));             // 32 KB
  half_t* packIn1 = (half_t*)carve(2 * 16 * 512 * sizeof(half_t));
  half_t* packX   = (half_t*)carve(KDIRS * 4 * 3 * 512 * sizeof(half_t));      // 48 KB
  half_t* packO0  = (half_t*)carve(4 * 4 * 512 * sizeof(half_t));              // 16 KB
  half_t* packO1  = (half_t*)carve(4 * 4 * 512 * sizeof(half_t));
  float*  xin0    = (float*)carve((size_t)D_INNER * LSEQ * sizeof(float));     // 2 MB
  float*  xin1    = (float*)carve((size_t)D_INNER * LSEQ * sizeof(float));
  float*  z0      = (float*)carve((size_t)LSEQ * D_INNER * sizeof(float));     // 2 MB
  float*  z1      = (float*)carve((size_t)LSEQ * D_INNER * sizeof(float));
  float*  xc0     = (float*)carve((size_t)D_INNER * LSEQ * sizeof(float));     // 2 MB
  float*  xc1     = (float*)carve((size_t)D_INNER * LSEQ * sizeof(float));
  half_t* xs0T    = (half_t*)carve((size_t)KDIRS * LSEQ * D_INNER * sizeof(half_t)); // 4 MB
  float*  xdbl    = (float*)carve((size_t)KDIRS * LSEQ * 48 * sizeof(float));  // 3 MB
  float*  delta   = (float*)carve((size_t)2 * KDIRS * D_INNER * LSEQ * sizeof(float)); // 16 MB
  float*  outy    = (float*)carve((size_t)2 * KDIRS * D_INNER * LSEQ * sizeof(float)); // 16 MB
  float*  ym      = (float*)carve((size_t)2 * LSEQ * D_INNER * sizeof(float)); // 4 MB
  half_t* gy      = (half_t*)carve((size_t)2 * LSEQ * D_INNER * sizeof(half_t)); // 2 MB

  const int TB = 256;

  // 1) convert activations to f16
  cvt_x_kernel<<<cdiv(2 * LSEQ * D_MODEL, TB), TB, 0, stream>>>(x0, x1, xh);

  // 2) pack all weight B-operands into WMMA fragment layout
  pack_b_kernel<<<cdiv(2 * 16 * 512, TB), TB, 0, stream>>>(in_w0, packIn0, 64, 256, 2, 16);
  pack_b_kernel<<<cdiv(2 * 16 * 512, TB), TB, 0, stream>>>(in_w1, packIn1, 64, 256, 2, 16);
  for (int k = 0; k < KDIRS; ++k)
    pack_b_kernel<<<cdiv(4 * 3 * 512, TB), TB, 0, stream>>>(
        xproj_w0 + (size_t)k * (DT_RANK + 2 * D_STATE) * D_INNER,
        packX + (size_t)k * 4 * 3 * 512, 128, 36, 4, 3);
  pack_b_kernel<<<cdiv(4 * 4 * 512, TB), TB, 0, stream>>>(out_w0, packO0, 128, 64, 4, 4);
  pack_b_kernel<<<cdiv(4 * 4 * 512, TB), TB, 0, stream>>>(out_w1, packO1, 128, 64, 4, 4);

  // 3) in_proj GEMMs (WMMA), split into conv input (NCHW) and gate z
  inproj_gemm_kernel<<<cdiv(256 * 16 * 32, TB), TB, 0, stream>>>(xh, packIn0, xin0, z0);
  inproj_gemm_kernel<<<cdiv(256 * 16 * 32, TB), TB, 0, stream>>>(
      xh + (size_t)LSEQ * D_MODEL, packIn1, xin1, z1);

  // 4) depthwise conv + SiLU
  dwconv_silu_kernel<<<cdiv(D_INNER * LSEQ, TB), TB, 0, stream>>>(xin0, conv0_w, conv0_b, xc0);
  dwconv_silu_kernel<<<cdiv(D_INNER * LSEQ, TB), TB, 0, stream>>>(xin1, conv1_w, conv1_b, xc1);

  // 5) build 4-direction xs0^T (f16) for the x_proj GEMM
  pack_xsT_kernel<<<cdiv(KDIRS * LSEQ * D_INNER, TB), TB, 0, stream>>>(xc0, xs0T);

  // 6) x_proj GEMM (WMMA): dt_rank/B/C streams for all 4 directions
  xdbl_gemm_kernel<<<cdiv(KDIRS * 256 * 3 * 32, TB), TB, 0, stream>>>(xs0T, packX, xdbl);

  // 7) delta = softplus(dt_w @ dts_r + dt_bias) for both parameter sets
  delta_kernel<<<cdiv(2 * KDIRS * D_INNER * LSEQ, TB), TB, 0, stream>>>(
      xdbl, dt_w0, dt_w1, dt_b0, dt_b1, delta);

  // 8) selective scans (cross-branch parameters), 512 waves
  scan_kernel<<<cdiv(512 * 32, TB), TB, 0, stream>>>(
      delta, xdbl, xc0, xc1, A_logs0, A_logs1, Ds0, Ds1, outy);

  // 9) merge directions
  merge_kernel<<<cdiv(2 * LSEQ * D_INNER, TB), TB, 0, stream>>>(outy, ym);

  // 10) layernorm + SiLU gate -> f16
  ln_gate_kernel<<<cdiv(2 * LSEQ * 32, TB), TB, 0, stream>>>(
      ym, z0, z1, ln0_g, ln0_b, ln1_g, ln1_b, gy);

  // 11) out_proj GEMM (WMMA) -> d_out (out0 then out1, f32)
  outproj_gemm_kernel<<<cdiv(2 * 256 * 4 * 32, TB), TB, 0, stream>>>(gy, packO0 /*base*/, out);
}
